// PreStackedMoEModel_22625887716001
// MI455X (gfx1250) — matrix operand order repaired and checked by reading, compiled, not run on hardware
//
#include <hip/hip_runtime.h>

typedef __attribute__((ext_vector_type(16))) __bf16         v16bf;
typedef __attribute__((ext_vector_type(8)))  float          v8f;
typedef __attribute__((ext_vector_type(8)))  unsigned short us8;
typedef __attribute__((ext_vector_type(16))) unsigned short us16;
typedef __attribute__((ext_vector_type(4)))  int            v4i;

#define T_TOK 8192
#define H_DIM 2048
#define I_DIM 1408
#define N_EXP 8
#define LDA   40           // padded LDS row stride (ushorts)
#define TBUF  (64 * LDA)   // one LDS tile buffer (64 rows x 32 k, padded)

// ---- workspace layout (bytes) ----
#define WS_COUNTS 0
#define WS_LIST   1024
#define WS_WGT    (WS_LIST + N_EXP * T_TOK * 4)
#define WS_XB     (((WS_WGT + N_EXP * T_TOK * 4) + 4095) & ~4095)
#define WS_HB     (WS_XB + T_TOK * H_DIM * 2)
#define WS_WGB    (WS_HB + T_TOK * I_DIM * 2)
#define WS_WDB    (WS_WGB + 2 * I_DIM * H_DIM * 2)
// total ~71 MB

#if __has_builtin(__builtin_amdgcn_global_load_async_to_lds_b128) && \
    __has_builtin(__builtin_amdgcn_s_wait_asynccnt)
#define HAVE_ASYNC 1
#else
#define HAVE_ASYNC 0
#endif

typedef __attribute__((address_space(1))) v4i* g_v4i_ptr;
typedef __attribute__((address_space(3))) v4i* l_v4i_ptr;

__device__ __forceinline__ void tile_copy16(const unsigned short* g, unsigned short* l) {
#if HAVE_ASYNC
  __builtin_amdgcn_global_load_async_to_lds_b128((g_v4i_ptr)g, (l_v4i_ptr)l, 0, 0);
#else
  *(us8*)l = *(const us8*)g;
#endif
}
__device__ __forceinline__ void tile_copy_wait() {
#if HAVE_ASYNC
  __builtin_amdgcn_s_wait_asynccnt(0);
#endif
}

__device__ __forceinline__ unsigned short f2bf(float f) {
  unsigned u = __builtin_bit_cast(unsigned, f);
  u += 0x7FFFu + ((u >> 16) & 1u);
  return (unsigned short)(u >> 16);
}
__device__ __forceinline__ unsigned pack2(float a, float b) {
#if __has_builtin(__builtin_amdgcn_cvt_pk_bf16_f32)
  return __builtin_bit_cast(unsigned, __builtin_amdgcn_cvt_pk_bf16_f32(a, b));
#else
  return (unsigned)f2bf(a) | ((unsigned)f2bf(b) << 16);
#endif
}
__device__ __forceinline__ float fast_rcp(float x) {
#if __has_builtin(__builtin_amdgcn_rcpf)
  return __builtin_amdgcn_rcpf(x);
#else
  return 1.0f / x;
#endif
}

// A fragment (16-bit, 16x32, ISA 7.12.2)
__device__ __forceinline__ v16bf ld_frag_a(const unsigned short* s, int row, int hi) {
  const us8 lo = *(const us8*)(s + row * LDA + hi * 8);
  const us8 hi8 = *(const us8*)(s + row * LDA + 16 + hi * 8);
  us16 c = __builtin_shufflevector(lo, hi8, 0,1,2,3,4,5,6,7,8,9,10,11,12,13,14,15);
  return __builtin_bit_cast(v16bf, c);
}
// B fragment (16-bit, 32x16): lane = column
__device__ __forceinline__ v16bf ld_frag_b(const unsigned short* s, int col, int hi) {
  const us8 lo = *(const us8*)(s + col * LDA + hi * 8);
  const us8 hi8 = *(const us8*)(s + col * LDA + 16 + hi * 8);
  us16 c = __builtin_shufflevector(lo, hi8, 0,1,2,3,4,5,6,7,8,9,10,11,12,13,14,15);
  return __builtin_bit_cast(v16bf, c);
}
__device__ __forceinline__ v8f do_wmma(v16bf a, v16bf b, v8f c) {
  return __builtin_amdgcn_wmma_f32_16x16x32_bf16(false, a, false, b, (short)0, c, false, false);
}

__global__ void k_init(float* __restrict__ out, int* __restrict__ counts, long n) {
  long i = (long)blockIdx.x * blockDim.x + threadIdx.x;
  long stride = (long)gridDim.x * blockDim.x;
  for (long j = i; j < n; j += stride) out[j] = 0.0f;
  if (i < N_EXP) counts[i] = 0;
}

__global__ void k_cvt(const float* __restrict__ src, unsigned short* __restrict__ dst, long n) {
  long i = ((long)blockIdx.x * blockDim.x + threadIdx.x) * 8;
  long stride = (long)gridDim.x * blockDim.x * 8;
  for (; i < n; i += stride) {
    float4 a = *(const float4*)(src + i);
    float4 b = *(const float4*)(src + i + 4);
    uint4 p;
    p.x = pack2(a.x, a.y); p.y = pack2(a.z, a.w);
    p.z = pack2(b.x, b.y); p.w = pack2(b.z, b.w);
    *(uint4*)(dst + i) = p;
  }
}

// one wave32 per token: logits -> softmax -> top2 -> append to expert lists
__global__ void __launch_bounds__(256) k_gate(const float* __restrict__ x,
                                              const float* __restrict__ gw,
                                              int* __restrict__ counts,
                                              int* __restrict__ list,
                                              float* __restrict__ wgt) {
  int lane = threadIdx.x & 31;
  int wave = threadIdx.x >> 5;
  int t = blockIdx.x * 8 + wave;
  float acc[N_EXP];
#pragma unroll
  for (int e = 0; e < N_EXP; ++e) acc[e] = 0.0f;
  const float* xr = x + (long)t * H_DIM;
  for (int k = lane; k < H_DIM; k += 32) {
    float xv = xr[k];
#pragma unroll
    for (int e = 0; e < N_EXP; ++e) acc[e] += xv * gw[e * H_DIM + k];
  }
#pragma unroll
  for (int e = 0; e < N_EXP; ++e) {
    float v = acc[e];
#pragma unroll
    for (int m = 16; m >= 1; m >>= 1) v += __shfl_xor(v, m, 32);
    acc[e] = v;
  }
  if (lane == 0) {
    float mx = acc[0];
#pragma unroll
    for (int e = 1; e < N_EXP; ++e) mx = fmaxf(mx, acc[e]);
    float p[N_EXP];
#pragma unroll
    for (int e = 0; e < N_EXP; ++e) p[e] = __expf(acc[e] - mx);
    int i0 = 0;
#pragma unroll
    for (int e = 1; e < N_EXP; ++e) if (p[e] > p[i0]) i0 = e;
    int i1 = (i0 == 0) ? 1 : 0;
#pragma unroll
    for (int e = 0; e < N_EXP; ++e) if (e != i0 && p[e] > p[i1]) i1 = e;
    float s = fast_rcp(p[i0] + p[i1]);
    int s0 = atomicAdd(&counts[i0], 1);
    list[i0 * T_TOK + s0] = t;  wgt[i0 * T_TOK + s0] = p[i0] * s;
    int s1 = atomicAdd(&counts[i1], 1);
    list[i1 * T_TOK + s1] = t;  wgt[i1 * T_TOK + s1] = p[i1] * s;
  }
}

// h[slot,n] = silu(x@Wg^T)*(x@Wu^T); M=64, N=64, double-buffered LDS, K=H_DIM
__global__ void __launch_bounds__(256) k_gateup(const unsigned short* __restrict__ xb,
                                                const unsigned short* __restrict__ wgb,
                                                const int* __restrict__ counts,
                                                const int* __restrict__ list,
                                                unsigned short* __restrict__ hbuf,
                                                int e) {
  __shared__ unsigned short sA[2 * TBUF];
  __shared__ unsigned short sG[2 * TBUF];
  __shared__ unsigned short sU[2 * TBUF];
  __shared__ int sTok[64];
  int cnt = counts[e];
  int bm = blockIdx.x, bn = blockIdx.y;
  if (bm * 64 >= cnt) return;
  int tid = threadIdx.x;
  if (tid < 64) {
    int s = bm * 64 + tid;
    sTok[tid] = (s < cnt) ? list[e * T_TOK + s] : list[e * T_TOK];
  }
  __syncthreads();
  int lane = tid & 31, wave = tid >> 5;
  int wm = wave >> 2, wn = wave & 3;      // waves 2 (M) x 4 (N)
  int mr = lane & 15, hi = lane >> 4;
  int row = tid >> 2, kq = (tid & 3) * 8; // copy: 64 rows x 32 k, 8 bf16/thread/tile
  const int lofs = row * LDA + kq;
  const long aofs = (long)sTok[row] * H_DIM + kq;
  const long gofs = (long)(bn * 64 + row) * H_DIM + kq;
  const long uofs = (long)(I_DIM + bn * 64 + row) * H_DIM + kq;
  v8f ag0 = {}, ag1 = {}, au0 = {}, au1 = {};

  // prologue: tile 0 -> buffer 0
  tile_copy16(xb + aofs, sA + lofs);
  tile_copy16(wgb + gofs, sG + lofs);
  tile_copy16(wgb + uofs, sU + lofs);
  tile_copy_wait();
  __syncthreads();

  int cur = 0;
  for (int kk = 0; kk < H_DIM - 32; kk += 32) {
    int nxt = cur ^ 1;
    // prefetch tile kk+32 into the other buffer
    tile_copy16(xb + aofs + kk + 32, sA + nxt * TBUF + lofs);
    tile_copy16(wgb + gofs + kk + 32, sG + nxt * TBUF + lofs);
    tile_copy16(wgb + uofs + kk + 32, sU + nxt * TBUF + lofs);
    // compute on current buffer
    const unsigned short* cA = sA + cur * TBUF;
    const unsigned short* cG = sG + cur * TBUF;
    const unsigned short* cU = sU + cur * TBUF;
    v16bf a0 = ld_frag_a(cA, wm * 32 + mr, hi);
    v16bf a1 = ld_frag_a(cA, wm * 32 + 16 + mr, hi);
    v16bf bg = ld_frag_b(cG, wn * 16 + mr, hi);
    v16bf bu = ld_frag_b(cU, wn * 16 + mr, hi);
    ag0 = do_wmma(a0, bg, ag0);
    ag1 = do_wmma(a1, bg, ag1);
    au0 = do_wmma(a0, bu, au0);
    au1 = do_wmma(a1, bu, au1);
    tile_copy_wait();
    __syncthreads();
    cur = nxt;
  }
  {  // last tile
    const unsigned short* cA = sA + cur * TBUF;
    const unsigned short* cG = sG + cur * TBUF;
    const unsigned short* cU = sU + cur * TBUF;
    v16bf a0 = ld_frag_a(cA, wm * 32 + mr, hi);
    v16bf a1 = ld_frag_a(cA, wm * 32 + 16 + mr, hi);
    v16bf bg = ld_frag_b(cG, wn * 16 + mr, hi);
    v16bf bu = ld_frag_b(cU, wn * 16 + mr, hi);
    ag0 = do_wmma(a0, bg, ag0);
    ag1 = do_wmma(a1, bg, ag1);
    au0 = do_wmma(a0, bu, au0);
    au1 = do_wmma(a1, bu, au1);
  }
  int n = bn * 64 + wn * 16 + mr;
#pragma unroll
  for (int f = 0; f < 2; ++f) {
#pragma unroll
    for (int r = 0; r < 8; ++r) {
      float g = f ? ag1[r] : ag0[r];
      float u = f ? au1[r] : au0[r];
      float hv = g * fast_rcp(1.0f + __expf(-g)) * u;   // silu(g)*u via v_rcp_f32
      int m = wm * 32 + f * 16 + r + hi * 8;
      hbuf[(long)(bm * 64 + m) * I_DIM + n] = f2bf(hv);
    }
  }
}

// out[token,n] += w[slot] * (h[slot] @ down^T); experts serialized on stream -> plain RMW
__global__ void __launch_bounds__(256) k_down(const unsigned short* __restrict__ hbuf,
                                              const unsigned short* __restrict__ wdb,
                                              const int* __restrict__ counts,
                                              const int* __restrict__ list,
                                              const float* __restrict__ wgt,
                                              float* __restrict__ out,
                                              int e) {
  __shared__ unsigned short sA[2 * TBUF];
  __shared__ unsigned short sB[2 * TBUF];
  __shared__ int sTok[64];
  __shared__ float sW[64];
  int cnt = counts[e];
  int bm = blockIdx.x, bn = blockIdx.y;
  if (bm * 64 >= cnt) return;
  int tid = threadIdx.x;
  if (tid < 64) {
    int s = bm * 64 + tid;
    sTok[tid] = (s < cnt) ? list[e * T_TOK + s] : 0;
    sW[tid]   = (s < cnt) ? wgt[e * T_TOK + s] : 0.0f;
  }
  __syncthreads();
  int lane = tid & 31, wave = tid >> 5;
  int wm = wave >> 2, wn = wave & 3;
  int mr = lane & 15, hi = lane >> 4;
  int row = tid >> 2, kq = (tid & 3) * 8;
  const int lofs = row * LDA + kq;
  const long aofs = (long)(bm * 64 + row) * I_DIM + kq;
  const long bofs = (long)(bn * 64 + row) * I_DIM + kq;
  v8f acc0 = {}, acc1 = {};

  tile_copy16(hbuf + aofs, sA + lofs);
  tile_copy16(wdb + bofs, sB + lofs);
  tile_copy_wait();
  __syncthreads();

  int cur = 0;
  for (int kk = 0; kk < I_DIM - 32; kk += 32) {
    int nxt = cur ^ 1;
    tile_copy16(hbuf + aofs + kk + 32, sA + nxt * TBUF + lofs);
    tile_copy16(wdb + bofs + kk + 32, sB + nxt * TBUF + lofs);
    const unsigned short* cA = sA + cur * TBUF;
    const unsigned short* cB = sB + cur * TBUF;
    v16bf a0 = ld_frag_a(cA, wm * 32 + mr, hi);
    v16bf a1 = ld_frag_a(cA, wm * 32 + 16 + mr, hi);
    v16bf b  = ld_frag_b(cB, wn * 16 + mr, hi);
    acc0 = do_wmma(a0, b, acc0);
    acc1 = do_wmma(a1, b, acc1);
    tile_copy_wait();
    __syncthreads();
    cur = nxt;
  }
  {
    const unsigned short* cA = sA + cur * TBUF;
    const unsigned short* cB = sB + cur * TBUF;
    v16bf a0 = ld_frag_a(cA, wm * 32 + mr, hi);
    v16bf a1 = ld_frag_a(cA, wm * 32 + 16 + mr, hi);
    v16bf b  = ld_frag_b(cB, wn * 16 + mr, hi);
    acc0 = do_wmma(a0, b, acc0);
    acc1 = do_wmma(a1, b, acc1);
  }
  int n = bn * 64 + wn * 16 + mr;
#pragma unroll
  for (int f = 0; f < 2; ++f) {
#pragma unroll
    for (int r = 0; r < 8; ++r) {
      int m = wm * 32 + f * 16 + r + hi * 8;
      int slot = bm * 64 + m;
      if (slot < cnt) {
        float v = f ? acc1[r] : acc0[r];
        out[(long)sTok[m] * H_DIM + n] += v * sW[m];
      }
    }
  }
}

extern "C" void kernel_launch(void* const* d_in, const int* in_sizes, int n_in,
                              void* d_out, int out_size, void* d_ws, size_t ws_size,
                              hipStream_t stream) {
  (void)in_sizes; (void)n_in; (void)out_size; (void)ws_size;
  const float* x   = (const float*)d_in[0];
  const float* gw  = (const float*)d_in[1];
  const float* gup = (const float*)d_in[2];
  const float* dwn = (const float*)d_in[3];
  float* out = (float*)d_out;
  char* ws = (char*)d_ws;
  int*            counts = (int*)(ws + WS_COUNTS);
  int*            list   = (int*)(ws + WS_LIST);
  float*          wgt    = (float*)(ws + WS_WGT);
  unsigned short* xb     = (unsigned short*)(ws + WS_XB);
  unsigned short* hbuf   = (unsigned short*)(ws + WS_HB);
  unsigned short* wgb    = (unsigned short*)(ws + WS_WGB);
  unsigned short* wdb    = (unsigned short*)(ws + WS_WDB);

  k_init<<<2048, 256, 0, stream>>>(out, counts, (long)T_TOK * H_DIM);
  k_cvt<<<2048, 256, 0, stream>>>(x, xb, (long)T_TOK * H_DIM);
  k_gate<<<T_TOK / 8, 256, 0, stream>>>(x, gw, counts, list, wgt);
  for (int e = 0; e < N_EXP; ++e) {
    k_cvt<<<1024, 256, 0, stream>>>(gup + (long)e * 2 * I_DIM * H_DIM, wgb,
                                    (long)2 * I_DIM * H_DIM);
    k_cvt<<<1024, 256, 0, stream>>>(dwn + (long)e * H_DIM * I_DIM, wdb,
                                    (long)H_DIM * I_DIM);
    dim3 g2(T_TOK / 64, I_DIM / 64);   // 128 x 22
    k_gateup<<<g2, 256, 0, stream>>>(xb, wgb, counts, list, hbuf, e);
    dim3 g3(T_TOK / 64, H_DIM / 64);   // 128 x 32
    k_down<<<g3, 256, 0, stream>>>(hbuf, wdb, counts, list, wgt, out, e);
  }
}
